// MoEFFN_10531259810700
// MI455X (gfx1250) — matrix operand order repaired and checked by reading, not compiled, not run on hardware
//
#include <hip/hip_runtime.h>
#include <math.h>

typedef _Float16 f16;
typedef __attribute__((ext_vector_type(16))) _Float16 v16h;
typedef __attribute__((ext_vector_type(8)))  _Float16 v8h;
typedef __attribute__((ext_vector_type(4)))  _Float16 v4h;
typedef __attribute__((ext_vector_type(8)))  float    v8f;
typedef __attribute__((ext_vector_type(4)))  unsigned int v4u;

#if __has_builtin(__builtin_amdgcn_tensor_load_to_lds)
#define HAS_TDM 1
typedef __attribute__((ext_vector_type(8))) int i32x8;
typedef __attribute__((ext_vector_type(4))) int i32x4;
#else
#define HAS_TDM 0
#endif

// Problem dimensions (fixed by the reference)
constexpr int   Bb  = 2, Tt = 2048, Hh = 1024;
constexpr int   Nn  = Bb * Tt;          // 4096 tokens
constexpr int   Ee  = 8, Ff = 2048;
constexpr int   CAP = 1280;             // int(1.25 * N*2/E)
constexpr int   ROWS = Ee * CAP;        // 10240 expert slots

union HV { v16h v; v8h h[2]; };

#if HAS_TDM
// Issue a TDM 2D tile load: tile_d1 rows x tile_d0 f16 elements from a
// row-major matrix with row pitch stride0_elems, into LDS at lds_off with
// 16B padding inserted after every 64B row chunk (=> 40-half LDS rows).
__device__ __forceinline__ void tdm_load_tile_2d(
    unsigned lds_off, const void* gptr,
    unsigned tensor_d0, unsigned stride0_elems,
    unsigned tile_d0, unsigned tile_d1) {
  unsigned long long ga = (unsigned long long)gptr;
  v4u g0;
  g0[0] = 1u;                                  // D# group0: count=1 (valid)
  g0[1] = lds_off;                             // lds_addr (bytes)
  g0[2] = (unsigned)ga;                        // global_addr[31:0]
  g0[3] = (unsigned)((ga >> 32) & 0x01FFFFFFu) // global_addr[56:32]
        | (2u << 30);                          // type = 2 ("image")
  i32x8 g1;
  g1[0] = (int)((1u << 16)    // data_size = 2 bytes
              | (1u << 20)    // pad_enable
              | (3u << 22)    // pad_interval: 8<<3 = 64 bytes
              | (3u << 25));  // pad_amount: 4 DWORDs = 16 bytes
  g1[1] = (int)((tensor_d0 & 0xFFFFu) << 16);                    // tdim0 lo
  g1[2] = (int)((tensor_d0 >> 16) | ((tile_d1 & 0xFFFFu) << 16));// tdim0 hi | tdim1 lo
  g1[3] = (int)((tile_d0 & 0xFFFFu) << 16);                      // tdim1 hi=0 | tile_dim0
  g1[4] = (int)(tile_d1 & 0xFFFFu);                              // tile_dim1 (tile_dim2=0)
  g1[5] = (int)stride0_elems;                                    // dim0 stride [31:0]
  g1[6] = 0;                                                     // stride hi | dim1 stride lo
  g1[7] = 0;
  i32x4 z4 = {0, 0, 0, 0};
  i32x8 z8 = {0, 0, 0, 0, 0, 0, 0, 0};
  __builtin_amdgcn_tensor_load_to_lds(g0, g1, z4, z4, z8, 0);
}
#endif

// ---------------------------------------------------------------- router ----
__global__ __launch_bounds__(256)
void router_k(const float* __restrict__ x, const float* __restrict__ Wr,
              float* __restrict__ gate, float* __restrict__ t2v,
              int* __restrict__ t2i) {
  int gtid = blockIdx.x * blockDim.x + threadIdx.x;
  int tok  = gtid >> 5;
  int lane = threadIdx.x & 31;
  if (tok >= Nn) return;
  const float* xr = x + (size_t)tok * Hh;
  float acc[Ee];
#pragma unroll
  for (int e = 0; e < Ee; ++e) acc[e] = 0.f;
  for (int k = lane; k < Hh; k += 32) {
    float xv = xr[k];
    const float* wr = Wr + (size_t)k * Ee;
#pragma unroll
    for (int e = 0; e < Ee; ++e) acc[e] += xv * wr[e];
  }
#pragma unroll
  for (int e = 0; e < Ee; ++e)
    for (int off = 16; off > 0; off >>= 1)
      acc[e] += __shfl_down(acc[e], off, 32);
  if (lane == 0) {
    float m = acc[0];
#pragma unroll
    for (int e = 1; e < Ee; ++e) m = fmaxf(m, acc[e]);
    float p[Ee]; float s = 0.f;
#pragma unroll
    for (int e = 0; e < Ee; ++e) { p[e] = expf(acc[e] - m); s += p[e]; }
    float inv = 1.0f / s;
#pragma unroll
    for (int e = 0; e < Ee; ++e) { p[e] *= inv; gate[(size_t)tok * Ee + e] = p[e]; }
    int   i0 = 0; float v0 = p[0];
#pragma unroll
    for (int e = 1; e < Ee; ++e) if (p[e] > v0) { v0 = p[e]; i0 = e; }
    int   i1 = (i0 == 0) ? 1 : 0; float v1 = p[i1];
#pragma unroll
    for (int e = 0; e < Ee; ++e)
      if (e != i0 && p[e] > v1) { v1 = p[e]; i1 = e; }
    t2v[2 * tok]     = v0; t2v[2 * tok + 1] = v1;
    t2i[2 * tok]     = i0; t2i[2 * tok + 1] = i1;
  }
}

// ---------------------------------------------------------------- scan ------
__global__ void scan_k(const int* __restrict__ t2i,
                       int* __restrict__ slot0, int* __restrict__ keep0,
                       int* __restrict__ slot1, int* __restrict__ keep1,
                       int* __restrict__ used) {
  int lane = threadIdx.x;
  __shared__ int u0s[Ee];
  if (lane < Ee) {
    int c0 = 0;
    for (int t = 0; t < Nn; ++t) {
      int e0 = t2i[2 * t];
      if (e0 == lane) {
        int p = c0++; int k = (p < CAP);
        slot0[t] = k ? lane * CAP + p : 0;
        keep0[t] = k;
      }
    }
    u0s[lane] = (c0 < CAP) ? c0 : CAP;
  }
  __syncthreads();
  if (lane < Ee) {
    int base = u0s[lane];
    int c1 = 0, u1 = 0;
    for (int t = 0; t < Nn; ++t) {
      int e1 = t2i[2 * t + 1];
      if (e1 == lane) {
        int p = base + c1++; int k = (p < CAP);
        slot1[t] = k ? lane * CAP + p : 0;
        keep1[t] = k;
        u1 += k;
      }
    }
    used[lane] = u0s[lane] + u1;
  }
}

// ---------------------------------------------------------------- loss ------
__global__ __launch_bounds__(256)
void loss_k(const float* __restrict__ gate, const int* __restrict__ used,
            float* __restrict__ out_loss) {
  __shared__ float part[256];
  int tid = threadIdx.x;
  int e = tid & 7, r = tid >> 3;
  float s = 0.f;
  for (int t = r; t < Nn; t += 32) s += gate[(size_t)t * Ee + e];
  part[tid] = s;
  __syncthreads();
  if (tid < Ee) {
    float tot = 0.f;
    for (int j = 0; j < 32; ++j) tot += part[j * Ee + tid]; // fixed order
    part[tid] = tot;
  }
  __syncthreads();
  if (tid == 0) {
    float loss = 0.f;
    for (int e2 = 0; e2 < Ee; ++e2) {
      float prob = part[e2] / (float)Nn;
      float fr   = fmaxf((float)used[e2], 1e-9f) / ((float)(2 * Nn) + 1e-9f);
      loss += prob * fr;
    }
    *out_loss = loss * (float)Ee;
  }
}

// ---------------------------------------------------------------- zero ------
__global__ void zero_k(v4u* __restrict__ p, long n4) {
  long i = (long)blockIdx.x * blockDim.x + threadIdx.x;
  if (i < n4) { v4u z = {0u, 0u, 0u, 0u}; p[i] = z; }
}

// ---------------------------------------------------------------- dispatch --
__global__ __launch_bounds__(256)
void dispatch_k(const float* __restrict__ x,
                const int* __restrict__ slot0, const int* __restrict__ keep0,
                const int* __restrict__ slot1, const int* __restrict__ keep1,
                f16* __restrict__ xe) {
  int t = blockIdx.x, tid = threadIdx.x;
  int h = tid * 4;
  const float4 v = *(const float4*)(x + (size_t)t * Hh + h);
  auto h01 = __builtin_amdgcn_cvt_pkrtz(v.x, v.y);
  auto h23 = __builtin_amdgcn_cvt_pkrtz(v.z, v.w);
  v4h hv = { (_Float16)h01[0], (_Float16)h01[1],
             (_Float16)h23[0], (_Float16)h23[1] };
  if (keep0[t]) *(v4h*)(xe + (size_t)slot0[t] * Hh + h) = hv;
  if (keep1[t]) *(v4h*)(xe + (size_t)slot1[t] * Hh + h) = hv;
}

// ---------------------------------------------------------------- GEMM 1 ----
// act[e] (CAP x F, f16) = SwiGLU( xe[e] (CAP x H, f16) @ W1[e] (H x 2F, f32) )
__global__ __launch_bounds__(256)
void gemm1_k(const f16* __restrict__ xe, const float* __restrict__ W1,
             f16* __restrict__ act) {
  constexpr int MT = CAP / 128;   // 10
  constexpr int NT = Ff / 64;     // 32
  constexpr int LD = Hh;          // A row pitch (K dim)
  constexpr int WLD = 2 * Ff;     // W1 row pitch
  int bid = blockIdx.x;
  int e   = bid / (MT * NT);
  int rem = bid % (MT * NT);
  int mt  = rem / NT, nt = rem % NT;
  const f16*   Abase = xe + (size_t)e * CAP * LD + (size_t)mt * 128 * LD;
  const float* Bw    = W1 + (size_t)e * Hh * WLD;
  int n0 = nt * 64;

  __shared__ f16 As[2][128][40];     // double-buffered, 40-half padded rows
  __shared__ f16 Bs[2][64][40];      // [branch][N][K] (B transposed)

  int tid  = threadIdx.x;
  int lane = tid & 31, w = tid >> 5;
  // Wave-uniform wave id in an SGPR: forces a *scalar* branch around the TDM
  // block (tensor ops ignore EXEC, so lane-predication would issue them from
  // every wave; a scalar branch makes waves 1..7 skip the issue entirely).
  int wid  = __builtin_amdgcn_readfirstlane(w);
  int wm = (w & 3) * 32;
  int wn = (w >> 2) * 32;
  int lr = lane & 15;
  int hi = lane >> 4;
  int kA = hi * 8;                   // A frag K-chunk base
  int kB = hi * 8;                   // B frag K-chunk base

  int ar  = tid >> 2;                // fallback A copy: row, +64 second pass
  int ac  = (tid & 3) * 8;
  int bn  = tid & 63;                // B loader: N within tile
  int bkc = (tid >> 6) * 4;          // B loader: K chunk base (0,4,8,12)

  v8f acc[2][2][2];
#pragma unroll
  for (int br = 0; br < 2; ++br)
    for (int mi = 0; mi < 2; ++mi)
      for (int ni = 0; ni < 2; ++ni)
        for (int d = 0; d < 8; ++d) acc[br][mi][ni][d] = 0.f;

#if HAS_TDM
  if (wid == 0)
    tdm_load_tile_2d((unsigned)(size_t)&As[0][0][0], Abase, LD, LD, 32, 128);
#endif

  for (int i = 0, k0 = 0; k0 < Hh; ++i, k0 += 32) {
    int cur = i & 1;
    // B tiles: 4 K-rows per thread along one column -> pack -> one b64 store
#pragma unroll
    for (int br = 0; br < 2; ++br) {
      const float* src = Bw + (size_t)k0 * WLD + n0 + br * Ff + bn;
#pragma unroll
      for (int p = 0; p < 2; ++p) {
        int kl = bkc + p * 16;
        const float* s2 = src + (size_t)kl * WLD;
        float f0 = s2[0];
        float f1 = s2[WLD];
        float f2 = s2[2 * WLD];
        float f3 = s2[3 * WLD];
        auto h01 = __builtin_amdgcn_cvt_pkrtz(f0, f1);
        auto h23 = __builtin_amdgcn_cvt_pkrtz(f2, f3);
        v4h hv = { (_Float16)h01[0], (_Float16)h01[1],
                   (_Float16)h23[0], (_Float16)h23[1] };
        *(v4h*)&Bs[br][bn][kl] = hv;
      }
    }
    if (k0 + 32 < Hh)
      __builtin_prefetch(Bw + (size_t)(k0 + 32 + bkc) * WLD + n0 + bn, 0, 1);

#if HAS_TDM
    if (wid == 0) __builtin_amdgcn_s_wait_tensorcnt(0);
    __syncthreads();
    if (wid == 0 && k0 + 32 < Hh)
      tdm_load_tile_2d((unsigned)(size_t)&As[cur ^ 1][0][0],
                       Abase + k0 + 32, LD, LD, 32, 128);
#else
    *(v4u*)&As[cur][ar][ac]      = *(const v4u*)(Abase + (size_t)ar * LD + k0 + ac);
    *(v4u*)&As[cur][ar + 64][ac] = *(const v4u*)(Abase + (size_t)(ar + 64) * LD + k0 + ac);
    __syncthreads();
#endif

    HV a[2], b[2][2];
#pragma unroll
    for (int mi = 0; mi < 2; ++mi) {
      int rr = wm + mi * 16 + lr;
      a[mi].h[0] = *(const v8h*)&As[cur][rr][kA];
      a[mi].h[1] = *(const v8h*)&As[cur][rr][kA + 16];
    }
#pragma unroll
    for (int br = 0; br < 2; ++br)
      for (int ni = 0; ni < 2; ++ni) {
        int cc = wn + ni * 16 + lr;
        b[br][ni].h[0] = *(const v8h*)&Bs[br][cc][kB];
        b[br][ni].h[1] = *(const v8h*)&Bs[br][cc][kB + 16];
      }
#pragma unroll
    for (int br = 0; br < 2; ++br)
      for (int mi = 0; mi < 2; ++mi)
        for (int ni = 0; ni < 2; ++ni)
          acc[br][mi][ni] = __builtin_amdgcn_wmma_f32_16x16x32_f16(
              false, a[mi].v, false, b[br][ni].v,
              (short)0, acc[br][mi][ni], false, false);
    __syncthreads();
  }

  size_t rowbase = (size_t)e * CAP + (size_t)mt * 128;
#pragma unroll
  for (int mi = 0; mi < 2; ++mi)
    for (int ni = 0; ni < 2; ++ni)
#pragma unroll
      for (int d = 0; d < 8; ++d) {
        int M  = wm + mi * 16 + d + hi * 8;
        int Fc = n0 + wn + ni * 16 + lr;
        float av = acc[0][mi][ni][d];
        float bv = acc[1][mi][ni][d];
        float sv = (av / (1.0f + __expf(-av))) * bv;   // silu(a) * b
        act[(rowbase + M) * (size_t)Ff + Fc] = (f16)sv;
      }
}

// ---------------------------------------------------------------- GEMM 2 ----
// obuf[e] (CAP x H, f32) = act[e] (CAP x F, f16) @ W2[e] (F x H, f32)
__global__ __launch_bounds__(256)
void gemm2_k(const f16* __restrict__ act, const float* __restrict__ W2,
             float* __restrict__ obuf) {
  constexpr int MT = CAP / 128;   // 10
  constexpr int NT = Hh / 64;     // 16
  constexpr int LD = Ff;          // A row pitch
  constexpr int WLD = Hh;         // W2 row pitch
  int bid = blockIdx.x;
  int e   = bid / (MT * NT);
  int rem = bid % (MT * NT);
  int mt  = rem / NT, nt = rem % NT;
  const f16*   Abase = act + (size_t)e * CAP * LD + (size_t)mt * 128 * LD;
  const float* Bw    = W2 + (size_t)e * Ff * WLD;
  int n0 = nt * 64;

  __shared__ f16 As[2][128][40];
  __shared__ f16 Bs[64][40];

  int tid  = threadIdx.x;
  int lane = tid & 31, w = tid >> 5;
  int wid  = __builtin_amdgcn_readfirstlane(w);   // uniform -> scalar branch
  int wm = (w & 3) * 32;
  int wn = (w >> 2) * 32;
  int lr = lane & 15;
  int hi = lane >> 4;
  int kA = hi * 8;
  int kB = hi * 8;

  int ar  = tid >> 2;
  int ac  = (tid & 3) * 8;
  int bn  = tid & 63;
  int bkc = (tid >> 6) * 4;

  v8f acc[2][2];
#pragma unroll
  for (int mi = 0; mi < 2; ++mi)
    for (int ni = 0; ni < 2; ++ni)
      for (int d = 0; d < 8; ++d) acc[mi][ni][d] = 0.f;

#if HAS_TDM
  if (wid == 0)
    tdm_load_tile_2d((unsigned)(size_t)&As[0][0][0], Abase, LD, LD, 32, 128);
#endif

  for (int i = 0, k0 = 0; k0 < Ff; ++i, k0 += 32) {
    int cur = i & 1;
    {
      const float* src = Bw + (size_t)k0 * WLD + n0 + bn;
#pragma unroll
      for (int p = 0; p < 2; ++p) {
        int kl = bkc + p * 16;
        const float* s2 = src + (size_t)kl * WLD;
        float f0 = s2[0];
        float f1 = s2[WLD];
        float f2 = s2[2 * WLD];
        float f3 = s2[3 * WLD];
        auto h01 = __builtin_amdgcn_cvt_pkrtz(f0, f1);
        auto h23 = __builtin_amdgcn_cvt_pkrtz(f2, f3);
        v4h hv = { (_Float16)h01[0], (_Float16)h01[1],
                   (_Float16)h23[0], (_Float16)h23[1] };
        *(v4h*)&Bs[bn][kl] = hv;
      }
    }
    if (k0 + 32 < Ff)
      __builtin_prefetch(Bw + (size_t)(k0 + 32 + bkc) * WLD + n0 + bn, 0, 1);

#if HAS_TDM
    if (wid == 0) __builtin_amdgcn_s_wait_tensorcnt(0);
    __syncthreads();
    if (wid == 0 && k0 + 32 < Ff)
      tdm_load_tile_2d((unsigned)(size_t)&As[cur ^ 1][0][0],
                       Abase + k0 + 32, LD, LD, 32, 128);
#else
    *(v4u*)&As[cur][ar][ac]      = *(const v4u*)(Abase + (size_t)ar * LD + k0 + ac);
    *(v4u*)&As[cur][ar + 64][ac] = *(const v4u*)(Abase + (size_t)(ar + 64) * LD + k0 + ac);
    __syncthreads();
#endif

    HV a[2], b[2];
#pragma unroll
    for (int mi = 0; mi < 2; ++mi) {
      int rr = wm + mi * 16 + lr;
      a[mi].h[0] = *(const v8h*)&As[cur][rr][kA];
      a[mi].h[1] = *(const v8h*)&As[cur][rr][kA + 16];
    }
#pragma unroll
    for (int ni = 0; ni < 2; ++ni) {
      int cc = wn + ni * 16 + lr;
      b[ni].h[0] = *(const v8h*)&Bs[cc][kB];
      b[ni].h[1] = *(const v8h*)&Bs[cc][kB + 16];
    }
#pragma unroll
    for (int mi = 0; mi < 2; ++mi)
      for (int ni = 0; ni < 2; ++ni)
        acc[mi][ni] = __builtin_amdgcn_wmma_f32_16x16x32_f16(
            false, a[mi].v, false, b[ni].v,
            (short)0, acc[mi][ni], false, false);
    __syncthreads();
  }

  size_t rowbase = (size_t)e * CAP + (size_t)mt * 128;
#pragma unroll
  for (int mi = 0; mi < 2; ++mi)
    for (int ni = 0; ni < 2; ++ni)
#pragma unroll
      for (int d = 0; d < 8; ++d) {
        int M = wm + mi * 16 + d + hi * 8;
        int c = n0 + wn + ni * 16 + lr;
        obuf[(rowbase + M) * (size_t)Hh + c] = acc[mi][ni][d];
      }
}

// ---------------------------------------------------------------- combine ---
__global__ __launch_bounds__(256)
void combine_k(const float* __restrict__ obuf,
               const int* __restrict__ slot0, const int* __restrict__ keep0,
               const int* __restrict__ slot1, const int* __restrict__ keep1,
               const float* __restrict__ t2v, float* __restrict__ y) {
  int t = blockIdx.x, tid = threadIdx.x;
  int h = tid * 4;
  float4 r = {0.f, 0.f, 0.f, 0.f};
  if (keep0[t]) {
    float v0 = t2v[2 * t];
    const float4 a = *(const float4*)(obuf + (size_t)slot0[t] * Hh + h);
    r.x += a.x * v0; r.y += a.y * v0; r.z += a.z * v0; r.w += a.w * v0;
  }
  if (keep1[t]) {
    float v1 = t2v[2 * t + 1];
    const float4 a = *(const float4*)(obuf + (size_t)slot1[t] * Hh + h);
    r.x += a.x * v1; r.y += a.y * v1; r.z += a.z * v1; r.w += a.w * v1;
  }
  *(float4*)(y + (size_t)t * Hh + h) = r;
}

// ---------------------------------------------------------------- launch ----
extern "C" void kernel_launch(void* const* d_in, const int* in_sizes, int n_in,
                              void* d_out, int out_size, void* d_ws, size_t ws_size,
                              hipStream_t stream) {
  (void)in_sizes; (void)n_in; (void)out_size; (void)ws_size;
  const float* x  = (const float*)d_in[0];
  const float* Wr = (const float*)d_in[1];
  const float* W1 = (const float*)d_in[2];
  const float* W2 = (const float*)d_in[3];
  float* y = (float*)d_out;

  char* ws = (char*)d_ws;
  auto alloc = [&](size_t bytes) -> char* {
    char* p = ws;
    ws += (bytes + 255) & ~(size_t)255;
    return p;
  };
  float* gate  = (float*)alloc((size_t)Nn * Ee * 4);
  float* t2v   = (float*)alloc((size_t)Nn * 2 * 4);
  int*   t2i   = (int*)  alloc((size_t)Nn * 2 * 4);
  int*   slot0 = (int*)  alloc((size_t)Nn * 4);
  int*   keep0 = (int*)  alloc((size_t)Nn * 4);
  int*   slot1 = (int*)  alloc((size_t)Nn * 4);
  int*   keep1 = (int*)  alloc((size_t)Nn * 4);
  int*   used  = (int*)  alloc(256);
  f16*   xe    = (f16*)  alloc((size_t)ROWS * Hh * 2);
  f16*   actb  = (f16*)  alloc((size_t)ROWS * Ff * 2);
  float* obuf  = (float*)alloc((size_t)ROWS * Hh * 4);

  router_k<<<Nn / 8, 256, 0, stream>>>(x, Wr, gate, t2v, t2i);
  scan_k<<<1, 32, 0, stream>>>(t2i, slot0, keep0, slot1, keep1, used);
  loss_k<<<1, 256, 0, stream>>>(gate, used, y + (size_t)Nn * Hh);

  long xe4 = (long)ROWS * Hh * 2 / 16;
  zero_k<<<(unsigned)((xe4 + 255) / 256), 256, 0, stream>>>((v4u*)xe, xe4);
  dispatch_k<<<Nn, 256, 0, stream>>>(x, slot0, keep0, slot1, keep1, xe);

  gemm1_k<<<Ee * (CAP / 128) * (Ff / 64), 256, 0, stream>>>(xe, W1, actb);
  gemm2_k<<<Ee * (CAP / 128) * (Hh / 64), 256, 0, stream>>>(actb, W2, obuf);

  combine_k<<<Nn, 256, 0, stream>>>(obuf, slot0, keep0, slot1, keep1, t2v, y);
}
